// MS_ESFG_63050119905938
// MI455X (gfx1250) — matrix operand order repaired and checked by reading, not compiled, not run on hardware
//
#include <hip/hip_runtime.h>
#include <hip/hip_bf16.h>
#include <cstdint>

#define DIM   128
#define NV    30000
#define NC    120000
#define NITER 4

typedef __attribute__((ext_vector_type(16))) __bf16 v16bf;
typedef __attribute__((ext_vector_type(8)))  __bf16 v8bf;
typedef __attribute__((ext_vector_type(8)))  float  v8f;

typedef unsigned int u32x4 __attribute__((ext_vector_type(4)));
typedef int          i32x4 __attribute__((ext_vector_type(4)));
typedef int          i32x8 __attribute__((ext_vector_type(8)));

#if __has_builtin(__builtin_amdgcn_tensor_load_to_lds)
#define HAVE_TDM 1
#else
#define HAVE_TDM 0
#endif

// ---------------------------------------------------------------------------
// Tensor Data Mover: DMA `ndw` dwords from global into LDS at byte offset
// lds_byte_addr.  D# built per CDNA5 ISA:
//  group0: count=1 | lds_addr | global_addr[56:0] | type=2
//  group1: data_size=4B, tensor_dim0=ndw, tensor_dim1=1, tile_dim0=ndw,
//          tile_dim1=1, tensor_dim0_stride=ndw
// Tracked with TENSORcnt.  Issue from ONE wave only (EXEC is ignored).
// ---------------------------------------------------------------------------
#if HAVE_TDM
__device__ __forceinline__ void tdm_load_dwords(const void* gsrc,
                                                unsigned lds_byte_addr,
                                                unsigned ndw)
{
    unsigned long long ga = (unsigned long long)(uintptr_t)gsrc;
    u32x4 g0 = { 1u,                                    // count=1, is_restore=0
                 lds_byte_addr,                          // lds_addr
                 (unsigned)(ga & 0xFFFFFFFFu),           // global_addr[31:0]
                 (unsigned)((ga >> 32) & 0x01FFFFFFu) | (2u << 30) }; // [56:32] | type=2
    i32x8 g1 = { (int)(2u << 16),                        // wg_mask=0, data_size=2 (4B)
                 (int)((ndw & 0xFFFFu) << 16),           // tensor_dim0[15:0]
                 (int)(((ndw >> 16) & 0xFFFFu) | (1u << 16)), // dim0[31:16], dim1=1
                 (int)((ndw & 0xFFFFu) << 16),           // tile_dim0 = ndw
                 1,                                      // tile_dim1=1, tile_dim2=0
                 (int)ndw,                               // dim0_stride[31:0]
                 0, 0 };
    i32x4 gz = { 0, 0, 0, 0 };
#if __clang_major__ >= 23
    i32x8 gz8 = { 0, 0, 0, 0, 0, 0, 0, 0 };
    __builtin_amdgcn_tensor_load_to_lds(g0, g1, gz, gz, gz8, 0);
#else
    __builtin_amdgcn_tensor_load_to_lds(g0, g1, gz, gz, 0);
#endif
}
#endif

// ---------------------------------------------------------------------------
// Pack an f32 weight matrix into bf16 WMMA B-fragments for
// v_wmma_f32_16x16x32_bf16.  B tile is 32(K) x 16(N).
// frag index fi = nt*KT + kt, then [lane][16 bf16].
// Lane layout: half=lane>>4, r=lane&15, col n = nt*16+r, elem e -> k=kt*32+(e<8 ? 8*half+e : 16+8*half+(e-8))
// ---------------------------------------------------------------------------
__global__ void pack_bfrag_kernel(const float* __restrict__ W,
                                  __bf16* __restrict__ dst,
                                  int K, int N, int ld, int transposed)
{
    int idx = blockIdx.x * blockDim.x + threadIdx.x;
    int KT = K >> 5, NT = N >> 4;
    int total = KT * NT * 32;
    if (idx >= total) return;
    int lane = idx & 31;
    int fi   = idx >> 5;
    int kt   = fi % KT;
    int nt   = fi / KT;
    int half = lane >> 4, r = lane & 15;
    int n = nt * 16 + r;
    __bf16* o = dst + (size_t)idx * 16;
#pragma unroll
    for (int e = 0; e < 16; ++e) {
        int k = kt * 32 + (e < 8 ? 8 * half + e : 16 + 8 * half + (e - 8));
        float w = transposed ? W[(size_t)n * ld + k] : W[(size_t)k * ld + n];
        o[e] = (__bf16)w;
    }
}

// A (16x32 bf16) fragment from a row-major f32 row pointer.
__device__ __forceinline__ v16bf load_afrag_f32(const float* rowptr, int kbase, int half)
{
    v16bf a;
#pragma unroll
    for (int j = 0; j < 8; ++j) {
        int kk = (j < 4) ? 2 * j : 16 + 2 * (j - 4);
        int k = kbase + kk + 8 * half;
        float2 f = *(const float2*)(rowptr + k);
        a[2 * j]     = (__bf16)f.x;
        a[2 * j + 1] = (__bf16)f.y;
    }
    return a;
}

// ---------------------------------------------------------------------------
// MLP: Y = relu(X @ W1 + b1) @ W2 + b2.  4 waves/block, 64 rows/block.
// Weights DMA'd into LDS via TDM (W1, then W2 reusing the same buffer).
// ---------------------------------------------------------------------------
#define WFRAG_DW 8192   // 4*8*32*16 bf16 = 16384 bf16 = 8192 dwords = 32 KB

__global__ __launch_bounds__(128) void mlp128_kernel(
    const float* __restrict__ X, int Nrows,
    const __bf16* __restrict__ W1pk, const float* __restrict__ b1,
    const __bf16* __restrict__ W2pk, const float* __restrict__ b2,
    float* __restrict__ Y)
{
    __shared__ __bf16 wbuf[WFRAG_DW * 2];         // 32 KB packed weight stage
    __shared__ __bf16 hbuf[4][16][DIM];           // 16 KB layer-1 activations

    int wave = threadIdx.x >> 5;
    int lane = threadIdx.x & 31;
    int half = lane >> 4, r = lane & 15;
    int rowBase = blockIdx.x * 64 + wave * 16;

    // kick off W1 DMA before touching A so it overlaps the activation loads
#if HAVE_TDM
    if (threadIdx.x == 0)
        tdm_load_dwords(W1pk, (unsigned)(uintptr_t)&wbuf[0], WFRAG_DW);
#else
    for (unsigned i = threadIdx.x; i < WFRAG_DW; i += 128)
        ((unsigned*)wbuf)[i] = ((const unsigned*)W1pk)[i];
#endif

    int row = rowBase + r;
    if (row >= Nrows) row = Nrows - 1;
    const float* xr = X + (size_t)row * DIM;

    v16bf a[4];
#pragma unroll
    for (int kt = 0; kt < 4; ++kt) a[kt] = load_afrag_f32(xr, kt * 32, half);

#if HAVE_TDM
    if (threadIdx.x == 0) __builtin_amdgcn_s_wait_tensorcnt(0);
#endif
    __syncthreads();

    // ----- layer 1: h = relu(X@W1 + b1), stage bf16 in LDS -----
#pragma unroll
    for (int nt = 0; nt < 8; ++nt) {
        float bv = b1[nt * 16 + r];
        v8f acc;
#pragma unroll
        for (int i = 0; i < 8; ++i) acc[i] = bv;
#pragma unroll
        for (int kt = 0; kt < 4; ++kt) {
            v16bf b = *(const v16bf*)(wbuf + ((size_t)(nt * 4 + kt) * 32 + lane) * 16);
            acc = __builtin_amdgcn_wmma_f32_16x16x32_bf16(false, a[kt], false, b,
                                                          (short)0, acc, false, false);
        }
#pragma unroll
        for (int i = 0; i < 8; ++i) {
            float v = acc[i] > 0.f ? acc[i] : 0.f;
            hbuf[wave][i + 8 * half][nt * 16 + r] = (__bf16)v;
        }
    }
    __syncthreads();                     // layer-1 reads of wbuf + hbuf writes done

    // ----- stage W2 into the same LDS buffer -----
#if HAVE_TDM
    if (threadIdx.x == 0) {
        tdm_load_dwords(W2pk, (unsigned)(uintptr_t)&wbuf[0], WFRAG_DW);
        __builtin_amdgcn_s_wait_tensorcnt(0);
    }
#else
    for (unsigned i = threadIdx.x; i < WFRAG_DW; i += 128)
        ((unsigned*)wbuf)[i] = ((const unsigned*)W2pk)[i];
#endif

    // rebuild A-frags from LDS while the DMA drains
    v16bf a2[4];
    {
        const __bf16* hrow = &hbuf[wave][r][0];
#pragma unroll
        for (int kt = 0; kt < 4; ++kt) {
            int k0 = kt * 32 + 8 * half;
            v8bf lo = *(const v8bf*)(hrow + k0);
            v8bf hi = *(const v8bf*)(hrow + k0 + 16);
            a2[kt] = __builtin_shufflevector(lo, hi, 0,1,2,3,4,5,6,7,8,9,10,11,12,13,14,15);
        }
    }
    __syncthreads();

    // ----- layer 2: Y = h@W2 + b2 -----
#pragma unroll
    for (int nt = 0; nt < 8; ++nt) {
        float bv = b2[nt * 16 + r];
        v8f acc;
#pragma unroll
        for (int i = 0; i < 8; ++i) acc[i] = bv;
#pragma unroll
        for (int kt = 0; kt < 4; ++kt) {
            v16bf b = *(const v16bf*)(wbuf + ((size_t)(nt * 4 + kt) * 32 + lane) * 16);
            acc = __builtin_amdgcn_wmma_f32_16x16x32_bf16(false, a2[kt], false, b,
                                                          (short)0, acc, false, false);
        }
#pragma unroll
        for (int i = 0; i < 8; ++i) {
            int orow = rowBase + i + 8 * half;
            if (orow < Nrows) Y[(size_t)orow * DIM + nt * 16 + r] = acc[i];
        }
    }
}

// ---------------------------------------------------------------------------
// Edge scatter: one wave per selected edge, fp32 global atomics.
// ---------------------------------------------------------------------------
__global__ __launch_bounds__(256) void scatter_kernel(
    const int* __restrict__ sel_edge,
    const int* __restrict__ v_edge, const int* __restrict__ c_edge,
    const float* __restrict__ msg_v,       // [V,128]
    const float* __restrict__ msg_c,       // [C,128]
    float* __restrict__ v2c,               // [C,256]
    float* __restrict__ c2v,               // [V,256]
    int nE, int colOff)
{
    int e = blockIdx.x * 8 + (threadIdx.x >> 5);
    int lane = threadIdx.x & 31;
    if (e >= nE) return;
    int ed  = sel_edge[e];
    int vid = v_edge[ed];
    int cid = c_edge[ed];

    float4 a = ((const float4*)(msg_v + (size_t)vid * DIM))[lane];
    float4 b = ((const float4*)(msg_c + (size_t)cid * DIM))[lane];
    float* dv2c = v2c + (size_t)cid * 256 + colOff + lane * 4;
    float* dc2v = c2v + (size_t)vid * 256 + colOff + lane * 4;
    atomicAdd(dv2c + 0, a.x); atomicAdd(dv2c + 1, a.y);
    atomicAdd(dv2c + 2, a.z); atomicAdd(dv2c + 3, a.w);
    atomicAdd(dc2v + 0, b.x); atomicAdd(dc2v + 1, b.y);
    atomicAdd(dc2v + 2, b.z); atomicAdd(dc2v + 3, b.w);
}

// ---------------------------------------------------------------------------
// GRU update, ir+hr / iz+hz fused into single WMMA accumulator chains.
// WihTpk: K=256 (KT=8), N=384 (NT=24) ; WhhTpk: K=128 (KT=4), NT=24.
// ---------------------------------------------------------------------------
__global__ __launch_bounds__(128) void gru_kernel(
    const float* __restrict__ Xin,         // [N,256]
    const float* __restrict__ Hin,         // [N,128]
    const __bf16* __restrict__ WihTpk,
    const __bf16* __restrict__ WhhTpk,
    const float* __restrict__ bih, const float* __restrict__ bhh,
    float* __restrict__ Hout, int Nrows)
{
    int wave = threadIdx.x >> 5;
    int lane = threadIdx.x & 31;
    int half = lane >> 4, r = lane & 15;
    int rowBase = blockIdx.x * 64 + wave * 16;
    int row = rowBase + r;
    if (row >= Nrows) row = Nrows - 1;

    const float* xr = Xin + (size_t)row * 256;
    const float* hr = Hin + (size_t)row * DIM;

    v16bf ax[8], ah[4];
#pragma unroll
    for (int kt = 0; kt < 8; ++kt) ax[kt] = load_afrag_f32(xr, kt * 32, half);
#pragma unroll
    for (int kt = 0; kt < 4; ++kt) ah[kt] = load_afrag_f32(hr, kt * 32, half);

#pragma unroll 2
    for (int nt = 0; nt < 8; ++nt) {                 // 8 output col tiles
        // prefetch next tile's fragment streams (global_prefetch_b8)
        if (nt + 1 < 8) {
            __builtin_prefetch(WihTpk + ((size_t)((nt + 1) * 8) * 32) * 16, 0, 0);
            __builtin_prefetch(WihTpk + ((size_t)((nt + 9) * 8) * 32) * 16, 0, 0);
            __builtin_prefetch(WihTpk + ((size_t)((nt + 17) * 8) * 32) * 16, 0, 0);
            __builtin_prefetch(WhhTpk + ((size_t)((nt + 1) * 4) * 32) * 16, 0, 0);
        }
        int col = nt * 16 + r;
        float br  = bih[col]           + bhh[col];
        float bz  = bih[128 + col]     + bhh[128 + col];
        float bnn = bih[256 + col];
        float bhn = bhh[256 + col];

        v8f accr, accz, accn, acchn;
#pragma unroll
        for (int i = 0; i < 8; ++i) { accr[i] = br; accz[i] = bz; accn[i] = bnn; acchn[i] = bhn; }

        // gate-r : ir (K=256) + hr (K=128) fused
#pragma unroll
        for (int kt = 0; kt < 8; ++kt) {
            v16bf b = *(const v16bf*)(WihTpk + ((size_t)(nt * 8 + kt) * 32 + lane) * 16);
            accr = __builtin_amdgcn_wmma_f32_16x16x32_bf16(false, ax[kt], false, b, (short)0, accr, false, false);
        }
#pragma unroll
        for (int kt = 0; kt < 4; ++kt) {
            v16bf b = *(const v16bf*)(WhhTpk + ((size_t)(nt * 4 + kt) * 32 + lane) * 16);
            accr = __builtin_amdgcn_wmma_f32_16x16x32_bf16(false, ah[kt], false, b, (short)0, accr, false, false);
        }
        // gate-z : col tile nt+8
#pragma unroll
        for (int kt = 0; kt < 8; ++kt) {
            v16bf b = *(const v16bf*)(WihTpk + ((size_t)((nt + 8) * 8 + kt) * 32 + lane) * 16);
            accz = __builtin_amdgcn_wmma_f32_16x16x32_bf16(false, ax[kt], false, b, (short)0, accz, false, false);
        }
#pragma unroll
        for (int kt = 0; kt < 4; ++kt) {
            v16bf b = *(const v16bf*)(WhhTpk + ((size_t)((nt + 8) * 4 + kt) * 32 + lane) * 16);
            accz = __builtin_amdgcn_wmma_f32_16x16x32_bf16(false, ah[kt], false, b, (short)0, accz, false, false);
        }
        // gate-n : inn (tile nt+16) and hn kept separate
#pragma unroll
        for (int kt = 0; kt < 8; ++kt) {
            v16bf b = *(const v16bf*)(WihTpk + ((size_t)((nt + 16) * 8 + kt) * 32 + lane) * 16);
            accn = __builtin_amdgcn_wmma_f32_16x16x32_bf16(false, ax[kt], false, b, (short)0, accn, false, false);
        }
#pragma unroll
        for (int kt = 0; kt < 4; ++kt) {
            v16bf b = *(const v16bf*)(WhhTpk + ((size_t)((nt + 16) * 4 + kt) * 32 + lane) * 16);
            acchn = __builtin_amdgcn_wmma_f32_16x16x32_bf16(false, ah[kt], false, b, (short)0, acchn, false, false);
        }

#pragma unroll
        for (int i = 0; i < 8; ++i) {
            int orow = rowBase + i + 8 * half;
            if (orow < Nrows) {
                float hprev = Hin[(size_t)orow * DIM + col];
                float rg = 1.f / (1.f + __expf(-accr[i]));
                float zg = 1.f / (1.f + __expf(-accz[i]));
                float ng = tanhf(accn[i] + rg * acchn[i]);
                Hout[(size_t)orow * DIM + col] = (1.f - zg) * ng + zg * hprev;
            }
        }
    }
}

__global__ void zero_f32_kernel(float* __restrict__ p, size_t n)
{
    size_t i = (size_t)blockIdx.x * blockDim.x + threadIdx.x;
    size_t stride = (size_t)gridDim.x * blockDim.x;
    for (; i < n; i += stride) p[i] = 0.f;
}

// ---------------------------------------------------------------------------
extern "C" void kernel_launch(void* const* d_in, const int* in_sizes, int n_in,
                              void* d_out, int out_size, void* d_ws, size_t ws_size,
                              hipStream_t stream)
{
    const int*   v_edge = (const int*)d_in[2];
    const int*   c_edge = (const int*)d_in[3];
    const int*   p_edge = (const int*)d_in[4];
    const int*   n_edge = (const int*)d_in[5];
    const float* v_emb  = (const float*)d_in[6];
    const float* c_emb  = (const float*)d_in[7];
    int EP = in_sizes[4];
    int EN = in_sizes[5];

    const float* mW1[4]; const float* mB1[4]; const float* mW2[4]; const float* mB2[4];
    for (int i = 0; i < 4; ++i) {
        mW1[i] = (const float*)d_in[8 + 4 * i + 0];
        mB1[i] = (const float*)d_in[8 + 4 * i + 1];
        mW2[i] = (const float*)d_in[8 + 4 * i + 2];
        mB2[i] = (const float*)d_in[8 + 4 * i + 3];
    }
    const float* WihC = (const float*)d_in[24]; const float* bihC = (const float*)d_in[25];
    const float* WhhC = (const float*)d_in[26]; const float* bhhC = (const float*)d_in[27];
    const float* WihV = (const float*)d_in[28]; const float* bihV = (const float*)d_in[29];
    const float* WhhV = (const float*)d_in[30]; const float* bhhV = (const float*)d_in[31];

    // ---- workspace carve ----
    char* ws = (char*)d_ws; size_t off = 0;
    auto take = [&](size_t bytes) -> void* {
        void* p = ws + off; off += (bytes + 255) & ~(size_t)255; return p;
    };
    __bf16* w1pk[4]; __bf16* w2pk[4];
    for (int i = 0; i < 4; ++i) { w1pk[i] = (__bf16*)take(4 * 8 * 32 * 16 * 2);
                                  w2pk[i] = (__bf16*)take(4 * 8 * 32 * 16 * 2); }
    __bf16* wihCpk = (__bf16*)take((size_t)8 * 24 * 32 * 16 * 2);
    __bf16* whhCpk = (__bf16*)take((size_t)4 * 24 * 32 * 16 * 2);
    __bf16* wihVpk = (__bf16*)take((size_t)8 * 24 * 32 * 16 * 2);
    __bf16* whhVpk = (__bf16*)take((size_t)4 * 24 * 32 * 16 * 2);
    float* mlp_v_pos = (float*)take((size_t)NV * DIM * 4);
    float* mlp_v_neg = (float*)take((size_t)NV * DIM * 4);
    float* mlp_c_pos = (float*)take((size_t)NC * DIM * 4);
    float* mlp_c_neg = (float*)take((size_t)NC * DIM * 4);
    float* v2c       = (float*)take((size_t)NC * 256 * 4);
    float* c2v       = (float*)take((size_t)NV * 256 * 4);

    // ---- pack weights into WMMA B-fragment layout (bf16) ----
    for (int i = 0; i < 4; ++i) {
        pack_bfrag_kernel<<<(4 * 8 * 32 + 255) / 256, 256, 0, stream>>>(mW1[i], w1pk[i], 128, 128, 128, 0);
        pack_bfrag_kernel<<<(4 * 8 * 32 + 255) / 256, 256, 0, stream>>>(mW2[i], w2pk[i], 128, 128, 128, 0);
    }
    pack_bfrag_kernel<<<(8 * 24 * 32 + 255) / 256, 256, 0, stream>>>(WihC, wihCpk, 256, 384, 256, 1);
    pack_bfrag_kernel<<<(4 * 24 * 32 + 255) / 256, 256, 0, stream>>>(WhhC, whhCpk, 128, 384, 128, 1);
    pack_bfrag_kernel<<<(8 * 24 * 32 + 255) / 256, 256, 0, stream>>>(WihV, wihVpk, 256, 384, 256, 1);
    pack_bfrag_kernel<<<(4 * 24 * 32 + 255) / 256, 256, 0, stream>>>(WhhV, whhVpk, 128, 384, 128, 1);

    // ---- output slabs: v_embs [5,V,128] then c_embs [5,C,128] ----
    float* outv = (float*)d_out;
    float* outc = outv + (size_t)(NITER + 1) * NV * DIM;
    hipMemcpyAsync(outv, v_emb, (size_t)NV * DIM * 4, hipMemcpyDeviceToDevice, stream);
    hipMemcpyAsync(outc, c_emb, (size_t)NC * DIM * 4, hipMemcpyDeviceToDevice, stream);

    const int gridV = (NV + 63) / 64;     // 469
    const int gridC = (NC + 63) / 64;     // 1875

    for (int t = 1; t <= NITER; ++t) {
        const float* vprev = outv + (size_t)(t - 1) * NV * DIM;
        const float* cprev = outc + (size_t)(t - 1) * NC * DIM;
        float* vnew = outv + (size_t)t * NV * DIM;
        float* cnew = outc + (size_t)t * NC * DIM;

        zero_f32_kernel<<<2048, 256, 0, stream>>>(v2c, (size_t)NC * 256);
        zero_f32_kernel<<<2048, 256, 0, stream>>>(c2v, (size_t)NV * 256);

        mlp128_kernel<<<gridV, 128, 0, stream>>>(vprev, NV, w1pk[0], mB1[0], w2pk[0], mB2[0], mlp_v_pos);
        mlp128_kernel<<<gridV, 128, 0, stream>>>(vprev, NV, w1pk[1], mB1[1], w2pk[1], mB2[1], mlp_v_neg);
        mlp128_kernel<<<gridC, 128, 0, stream>>>(cprev, NC, w1pk[2], mB1[2], w2pk[2], mB2[2], mlp_c_pos);
        mlp128_kernel<<<gridC, 128, 0, stream>>>(cprev, NC, w1pk[3], mB1[3], w2pk[3], mB2[3], mlp_c_neg);

        scatter_kernel<<<(EP + 7) / 8, 256, 0, stream>>>(p_edge, v_edge, c_edge,
                                                         mlp_v_pos, mlp_c_pos, v2c, c2v, EP, 0);
        scatter_kernel<<<(EN + 7) / 8, 256, 0, stream>>>(n_edge, v_edge, c_edge,
                                                         mlp_v_neg, mlp_c_neg, v2c, c2v, EN, 128);

        gru_kernel<<<gridC, 128, 0, stream>>>(v2c, cprev, wihCpk, whhCpk, bihC, bhhC, cnew, NC);
        gru_kernel<<<gridV, 128, 0, stream>>>(c2v, vprev, wihVpk, whhVpk, bihV, bhhV, vnew, NV);
    }
}
